// CuboidSelfAttention_40037685133384
// MI455X (gfx1250) — matrix operand order repaired and checked by reading, not compiled, not run on hardware
//
#include <hip/hip_runtime.h>
#include <stdint.h>

typedef __attribute__((ext_vector_type(16))) __bf16    v16bf;
typedef __attribute__((ext_vector_type(8)))  float     v8f;
typedef __attribute__((ext_vector_type(8)))  uint32_t  v8u;

#define NW    128      // total windows (B * 8 * 8)
#define NTOK  512      // tokens per window (T * WS * WS)
#define CC    256
#define NHEAD 8
#define DH    32
#define SCALE 0.17677669529663687f   // 1/sqrt(32)

// ---------- workspace layout (bytes) ----------
static constexpr size_t SZ_XW = (size_t)NW * NTOK * CC * 2;        // 33.5 MB bf16 windowed x
static constexpr size_t SZ_WQ = (size_t)768 * 256 * 2;             // qkv_w transposed bf16
static constexpr size_t SZ_WP = (size_t)256 * 256 * 2;             // proj_w transposed bf16
static constexpr size_t SZ_HB = (size_t)NW * NHEAD * NTOK * DH * 2;// per-head Q/K/V bf16
static constexpr size_t SZ_AO = (size_t)NW * NTOK * CC * 2;        // attention out bf16
static constexpr size_t OFF_XW = 0;
static constexpr size_t OFF_WQ = OFF_XW + SZ_XW;
static constexpr size_t OFF_WP = OFF_WQ + SZ_WQ;
static constexpr size_t OFF_Q  = OFF_WP + SZ_WP;
static constexpr size_t OFF_K  = OFF_Q  + SZ_HB;
static constexpr size_t OFF_V  = OFF_K  + SZ_HB;
static constexpr size_t OFF_AO = OFF_V  + SZ_HB;

__device__ __forceinline__ uint16_t f2bf(float f) {
  uint32_t u = __float_as_uint(f);
  return (uint16_t)((u + 0x7FFFu + ((u >> 16) & 1u)) >> 16);   // RNE
}

__device__ __forceinline__ v8u mk8(uint4 lo, uint4 hi) {
  v8u r;
  r[0]=lo.x; r[1]=lo.y; r[2]=lo.z; r[3]=lo.w;
  r[4]=hi.x; r[5]=hi.y; r[6]=hi.z; r[7]=hi.w;
  return r;
}

__device__ __forceinline__ v8f wmma_bf16(v8u a, v8u b, v8f c) {
  return __builtin_amdgcn_wmma_f32_16x16x32_bf16(
      false, __builtin_bit_cast(v16bf, a),
      false, __builtin_bit_cast(v16bf, b),
      (short)0, c, false, false);
}

__device__ __forceinline__ float hmax16(float v) {
  #pragma unroll
  for (int m = 1; m < 16; m <<= 1) v = fmaxf(v, __shfl_xor(v, m, 32));
  return v;
}
__device__ __forceinline__ float hsum16(float v) {
  #pragma unroll
  for (int m = 1; m < 16; m <<= 1) v += __shfl_xor(v, m, 32);
  return v;
}

// ---------------- kernel 1: transpose + bf16 weights ----------------
__global__ __launch_bounds__(256) void k_prep_w(const float* __restrict__ qw,
                                                const float* __restrict__ pw,
                                                uint16_t* __restrict__ wqT,
                                                uint16_t* __restrict__ wpT) {
  int idx = blockIdx.x * 256 + threadIdx.x;      // 768*256 + 256*256 = 262144 threads
  const int NQ = 768 * 256;
  if (idx < NQ) {
    int cout = idx >> 8, cin = idx & 255;        // wqT[cout][cin] = qw[cin][cout]
    wqT[idx] = f2bf(qw[cin * 768 + cout]);
  } else {
    int j = idx - NQ;
    int cout = j >> 8, cin = j & 255;
    wpT[j] = f2bf(pw[cin * 256 + cout]);
  }
}

// ---------------- kernel 2: window partition + bf16 ----------------
__global__ __launch_bounds__(256) void k_part(const float* __restrict__ x,
                                              uint16_t* __restrict__ xw) {
  size_t tid = (size_t)blockIdx.x * 256 + threadIdx.x;   // 4,194,304 threads, 4 ch each
  int c4  = tid & 63;
  int n   = (tid >> 6) & 511;
  int win = (int)(tid >> 15);
  int b = win >> 6, hi = (win >> 3) & 7, wi = win & 7;
  int t = n >> 6, ys = (n >> 3) & 7, xs = n & 7;
  int h = hi * 8 + ys, w = wi * 8 + xs;
  const float4 v = *(const float4*)(x + ((((size_t)(b * 8 + t) * 64 + h) * 64 + w) * 256 + c4 * 4));
  uint32_t lo = (uint32_t)f2bf(v.x) | ((uint32_t)f2bf(v.y) << 16);
  uint32_t hi2 = (uint32_t)f2bf(v.z) | ((uint32_t)f2bf(v.w) << 16);
  *(uint2*)(xw + ((size_t)win * NTOK + n) * CC + c4 * 4) = make_uint2(lo, hi2);
}

// ---------------- kernel 3: QKV GEMM (512x256 @ 256x768 per window) ----------------
// One wave computes a 16x64 output block: 8 K-steps x 4 N-tiles = 32 WMMA.
__global__ __launch_bounds__(256) void k_qkv(const uint16_t* __restrict__ xw,
                                             const uint16_t* __restrict__ wT,
                                             uint16_t* __restrict__ qb,
                                             uint16_t* __restrict__ kb,
                                             uint16_t* __restrict__ vt) {
  int wave = (blockIdx.x * 256 + threadIdx.x) >> 5;
  int lane = threadIdx.x & 31;
  int l16 = lane & 15, half = lane >> 4;
  int g   = wave % 12;             // column group of 64 (0..11)
  int mt  = (wave / 12) & 31;      // row tile (0..31)
  int win = wave / (12 * 32);
  const uint16_t* aBase = xw + ((size_t)win * NTOK + mt * 16 + l16) * CC;
  v8f acc[4] = {};
  #pragma unroll
  for (int kk = 0; kk < 8; ++kk) {
    const uint16_t* ap = aBase + kk * 32;
    v8u a = mk8(*(const uint4*)(ap + half * 8), *(const uint4*)(ap + 16 + half * 8));
    #pragma unroll
    for (int st = 0; st < 4; ++st) {
      int col = g * 64 + st * 16 + l16;
      const uint16_t* bp = wT + (size_t)col * CC + kk * 32 + half * 8;
      v8u b = mk8(*(const uint4*)bp, *(const uint4*)(bp + 16));
      acc[st] = wmma_bf16(a, b, acc[st]);
    }
  }
  #pragma unroll
  for (int st = 0; st < 4; ++st) {
    int cout = g * 64 + st * 16 + l16;
    int sec = cout >> 8, cc = cout & 255;        // 0=Q 1=K 2=V
    int head = cc >> 5, dh = cc & 31;
    size_t hb = (size_t)win * NHEAD + head;
    #pragma unroll
    for (int i = 0; i < 8; ++i) {
      int n = mt * 16 + i + 8 * half;
      uint16_t bv = f2bf(acc[st][i]);
      if (sec == 0)      qb[(hb * NTOK + n) * DH + dh] = bv;      // (win,head,n,dh)
      else if (sec == 1) kb[(hb * NTOK + n) * DH + dh] = bv;      // (win,head,n,dh)
      else               vt[(hb * DH + dh) * NTOK + n] = bv;      // (win,head,dh,n)
    }
  }
}

// ---------------- kernel 4: flash attention, one wave = 16 query rows ----------------
__global__ __launch_bounds__(256) void k_attn(const uint16_t* __restrict__ qb,
                                              const uint16_t* __restrict__ kb,
                                              const uint16_t* __restrict__ vt,
                                              uint16_t* __restrict__ aout) {
  __shared__ uint16_t lds[8][16 * 40];           // per-wave 16x32 bf16 P tile, 80B row stride
  int wv = threadIdx.x >> 5;
  int lane = threadIdx.x & 31;
  int l16 = lane & 15, half = lane >> 4;
  int gwave = blockIdx.x * 8 + wv;
  int mt = gwave & 31;                            // query row-block
  int wh = gwave >> 5;                            // (win,head) flat, 0..1023
  const uint16_t* qh = qb + (size_t)wh * NTOK * DH;
  const uint16_t* kh = kb + (size_t)wh * NTOK * DH;
  const uint16_t* vh = vt + (size_t)wh * DH * NTOK;
  // Q A-tile (16x32, K = dh = 32): lane row l16, K halves per ISA layout
  const uint16_t* qp = qh + (size_t)(mt * 16 + l16) * DH;
  v8u aq = mk8(*(const uint4*)(qp + half * 8), *(const uint4*)(qp + 16 + half * 8));
  float mrow[8], lrow[8];
  #pragma unroll
  for (int i = 0; i < 8; ++i) { mrow[i] = -__builtin_inff(); lrow[i] = 0.f; }
  v8f o0 = {}, o1 = {};
  uint16_t* lb = &lds[wv][0];
  for (int jp = 0; jp < 16; ++jp) {              // 32 key tokens per iteration
    int n0 = jp * 32;
    const uint16_t* kp0 = kh + (size_t)(n0 + l16) * DH + half * 8;
    const uint16_t* kp1 = kh + (size_t)(n0 + 16 + l16) * DH + half * 8;
    v8u bk0 = mk8(*(const uint4*)kp0, *(const uint4*)(kp0 + 16));
    v8u bk1 = mk8(*(const uint4*)kp1, *(const uint4*)(kp1 + 16));
    v8f z = {};
    v8f s0 = wmma_bf16(aq, bk0, z);              // S = Q K^T  (two 16x16 tiles)
    v8f s1 = wmma_bf16(aq, bk1, z);
    #pragma unroll
    for (int i = 0; i < 8; ++i) {                // online softmax per row
      float v0 = s0[i] * SCALE, v1 = s1[i] * SCALE;
      float tm = hmax16(fmaxf(v0, v1));
      float mn = fmaxf(mrow[i], tm);
      float corr = __expf(mrow[i] - mn);
      float p0 = __expf(v0 - mn), p1 = __expf(v1 - mn);
      lrow[i] = lrow[i] * corr + hsum16(p0 + p1);
      mrow[i] = mn;
      o0[i] *= corr; o1[i] *= corr;
      int r = i + 8 * half;                      // C-layout row -> LDS row-major bf16
      lb[r * 40 + l16]      = f2bf(p0);
      lb[r * 40 + 16 + l16] = f2bf(p1);
    }
    // reread P in A-matrix layout (16x32)
    const uint16_t* lp = lb + l16 * 40;
    v8u apm = mk8(*(const uint4*)(lp + half * 8), *(const uint4*)(lp + 16 + half * 8));
    // V B-tiles (K = 32 tokens, N = dh halves) from transposed V
    const uint16_t* vp0 = vh + (size_t)l16 * NTOK + n0 + half * 8;
    const uint16_t* vp1 = vh + (size_t)(16 + l16) * NTOK + n0 + half * 8;
    v8u bv0 = mk8(*(const uint4*)vp0, *(const uint4*)(vp0 + 16));
    v8u bv1 = mk8(*(const uint4*)vp1, *(const uint4*)(vp1 + 16));
    o0 = wmma_bf16(apm, bv0, o0);
    o1 = wmma_bf16(apm, bv1, o1);
  }
  int win = wh >> 3, head = wh & 7;
  #pragma unroll
  for (int i = 0; i < 8; ++i) {
    float inv = 1.f / lrow[i];
    int n = mt * 16 + i + 8 * half;
    size_t base = ((size_t)win * NTOK + n) * CC + head * DH;
    aout[base + l16]      = f2bf(o0[i] * inv);
    aout[base + 16 + l16] = f2bf(o1[i] * inv);
  }
}

// ---------------- kernel 5: output projection + bias + window reverse ----------------
__global__ __launch_bounds__(256) void k_proj(const uint16_t* __restrict__ aout,
                                              const uint16_t* __restrict__ pT,
                                              const float* __restrict__ pb,
                                              float* __restrict__ out) {
  int wave = (blockIdx.x * 256 + threadIdx.x) >> 5;
  int lane = threadIdx.x & 31;
  int l16 = lane & 15, half = lane >> 4;
  int g  = wave & 3;                              // column group of 64
  int mt = wave >> 2;                             // row tile over 65536 rows (0..4095)
  int rowbase = mt * 16;
  const uint16_t* aBase = aout + (size_t)(rowbase + l16) * CC;
  v8f acc[4] = {};
  #pragma unroll
  for (int kk = 0; kk < 8; ++kk) {
    const uint16_t* ap = aBase + kk * 32;
    v8u a = mk8(*(const uint4*)(ap + half * 8), *(const uint4*)(ap + 16 + half * 8));
    #pragma unroll
    for (int st = 0; st < 4; ++st) {
      int col = g * 64 + st * 16 + l16;
      const uint16_t* bp = pT + (size_t)col * CC + kk * 32 + half * 8;
      v8u b = mk8(*(const uint4*)bp, *(const uint4*)(bp + 16));
      acc[st] = wmma_bf16(a, b, acc[st]);
    }
  }
  int win = rowbase >> 9, nbase = rowbase & 511;
  int b_ = win >> 6, hi = (win >> 3) & 7, wi = win & 7;
  #pragma unroll
  for (int st = 0; st < 4; ++st) {
    int cout = g * 64 + st * 16 + l16;
    float bias = pb[cout];
    #pragma unroll
    for (int i = 0; i < 8; ++i) {
      int n = nbase + i + 8 * half;
      int t = n >> 6, ys = (n >> 3) & 7, xs = n & 7;
      int h = hi * 8 + ys, w = wi * 8 + xs;
      out[(((size_t)(b_ * 8 + t) * 64 + h) * 64 + w) * 256 + cout] = acc[st][i] + bias;
    }
  }
}

extern "C" void kernel_launch(void* const* d_in, const int* in_sizes, int n_in,
                              void* d_out, int out_size, void* d_ws, size_t ws_size,
                              hipStream_t stream) {
  (void)in_sizes; (void)n_in; (void)out_size; (void)ws_size;
  const float* x      = (const float*)d_in[0];
  const float* qkv_w  = (const float*)d_in[1];
  const float* proj_w = (const float*)d_in[2];
  const float* proj_b = (const float*)d_in[3];
  float* out = (float*)d_out;
  char* ws = (char*)d_ws;
  uint16_t* xw = (uint16_t*)(ws + OFF_XW);
  uint16_t* wq = (uint16_t*)(ws + OFF_WQ);
  uint16_t* wp = (uint16_t*)(ws + OFF_WP);
  uint16_t* qb = (uint16_t*)(ws + OFF_Q);
  uint16_t* kb = (uint16_t*)(ws + OFF_K);
  uint16_t* vt = (uint16_t*)(ws + OFF_V);
  uint16_t* ao = (uint16_t*)(ws + OFF_AO);

  k_prep_w<<<1024, 256, 0, stream>>>(qkv_w, proj_w, wq, wp);
  k_part  <<<16384, 256, 0, stream>>>(x, xw);
  k_qkv   <<<6144, 256, 0, stream>>>(xw, wq, qb, kb, vt);   // 49152 waves
  k_attn  <<<4096, 256, 0, stream>>>(qb, kb, vt, ao);       // 32768 waves
  k_proj  <<<2048, 256, 0, stream>>>(ao, wp, proj_b, out);  // 16384 waves
}
